// RNNClassifier_18322330485398
// MI455X (gfx1250) — matrix operand order repaired and checked by reading, not compiled, not run on hardware
//
#include <hip/hip_runtime.h>
#include <hip/hip_bf16.h>
#include <math.h>

// Problem constants (must match reference)
#define BB 256
#define TT 512
#define VV 4096
#define EE 256
#define HH 512
#define CC 10

#define LDH (HH + 8)   // padded LDS row stride (bf16 elems), 16B-aligned rows
#define LDE (EE + 8)

#define KCE (EE / 32)  // K-chunks for input projection (8)
#define KCH (HH / 32)  // K-chunks for recurrence (16)

typedef __attribute__((ext_vector_type(16))) __bf16 v16bf;
typedef __attribute__((ext_vector_type(8)))  __bf16 bf16x8;
typedef __attribute__((ext_vector_type(8)))  float  v8f;

// ---------------------------------------------------------------------------
// A-fragment swizzle: element (n,k) of row-major W[N][K] -> flat index such
// that fragment (ntile, kchunk) is 32 lanes x 16 bf16 (32B) contiguous, in the
// exact per-lane order of the WMMA 16-bit A operand (16x32, M=n, K=k):
//   lane = n%16 + 16*(seg&1),  elem = k%8 + 8*(seg>>1),  seg = (k%32)/8
// ---------------------------------------------------------------------------
__host__ __device__ __forceinline__ size_t swzA(int n, int k, int K) {
  const int ntile = n >> 4, lm = n & 15;
  const int kc = k >> 5, kin = k & 31;
  const int seg = kin >> 3, sub = kin & 7;
  const int lane = lm + ((seg & 1) << 4);
  const int elem = sub + ((seg >> 1) << 3);
  return ((size_t)(ntile * (K >> 5) + kc) * 32 + lane) * 16 + elem;
}

// ---------------------------------------------------------------------------
// Prologue: fp32 -> bf16 conversion; weights scattered into fragment layout.
// ---------------------------------------------------------------------------
__global__ __launch_bounds__(256) void cvt_kernel(
    const float* __restrict__ emb, const float* __restrict__ wih,
    const float* __restrict__ whh, const float* __restrict__ bih,
    const float* __restrict__ bhh,
    __bf16* __restrict__ emb_bf, __bf16* __restrict__ wih_sw,
    __bf16* __restrict__ whh_sw, float* __restrict__ bias_c) {
  int i = blockIdx.x * 256 + threadIdx.x;
  if (i < VV * EE) emb_bf[i] = (__bf16)emb[i];
  if (i < HH * EE) {
    const int n = i / EE, k = i % EE;
    wih_sw[swzA(n, k, EE)] = (__bf16)wih[i];
  }
  if (i < HH * HH) {
    const int n = i / HH, k = i % HH;
    whh_sw[swzA(n, k, HH)] = (__bf16)whh[i];
  }
  if (i < HH) bias_c[i] = bih[i] + bhh[i];
}

// Activation B-fragment: lane L -> batch row L%16, K values k0+8h..k0+8h+7 and
// k0+16+8h..k0+16+8h+7, h = (L>=16).  (two 16B LDS loads)
__device__ __forceinline__ v16bf load_act_frag(const __bf16* row, int k0,
                                               int hiB) {
  bf16x8 b0 = *(const bf16x8*)(row + k0 + (hiB >> 1));
  bf16x8 b1 = *(const bf16x8*)(row + k0 + (hiB >> 1) + 16);
  return __builtin_shufflevector(b0, b1, 0, 1, 2, 3, 4, 5, 6, 7,
                                 8, 9, 10, 11, 12, 13, 14, 15);
}

// Stage one 16-element segment of one embedding row into LDS (bf16).
__device__ __forceinline__ void stage_xe(const int* __restrict__ x,
                                         const __bf16* __restrict__ emb,
                                         __bf16* dst, int b0, int r, int cseg,
                                         int t) {
  const int idx = x[(b0 + r) * TT + t];
  const __bf16* src = emb + (size_t)idx * EE + cseg * 16;
  bf16x8 e0 = *(const bf16x8*)(src);
  bf16x8 e1 = *(const bf16x8*)(src + 8);
  *(bf16x8*)&dst[r * LDE + cseg * 16]     = e0;
  *(bf16x8*)&dst[r * LDE + cseg * 16 + 8] = e1;
}

// tanh(x) = 1 - 2/(exp(2x)+1): ~6 VALU/trans ops, exact at saturation,
// error well below bf16 storage rounding.
__device__ __forceinline__ float fast_tanh(float v) {
  const float e = __expf(2.0f * v);
  return 1.0f - __fdividef(2.0f, e + 1.0f);
}

// ---------------------------------------------------------------------------
// Fused RNN: one workgroup owns 16 batch rows for all 512 timesteps.
// 256 threads = 8 wave32; wave w computes hidden columns [w*64, w*64+64).
// Transposed WMMA: D(16 hid x 16 batch) = W-tile(16x32) x act^T(32x16) + D.
// Double-buffered h/xe in LDS -> exactly one barrier per timestep.
// ---------------------------------------------------------------------------
__global__ __launch_bounds__(256, 1) void rnn_kernel(
    const int* __restrict__ x,            // [B,T]
    const __bf16* __restrict__ emb,       // [V,E] bf16 row-major
    const __bf16* __restrict__ wih,       // [H,E] bf16, A-fragment swizzled
    const __bf16* __restrict__ whh,       // [H,H] bf16, A-fragment swizzled
    const float* __restrict__ bias,       // [H]  (b_ih+b_hh)
    const float* __restrict__ fcw,        // [C,H] fp32
    const float* __restrict__ fcb,        // [C]
    float* __restrict__ out) {            // [B,C]
  __shared__ __bf16 h_lds[2][16 * LDH];
  __shared__ __bf16 xe_lds[2][16 * LDE];

  const int tid  = threadIdx.x;
  const int lane = tid & 31;
  const int wave = tid >> 5;
  const int b0   = blockIdx.x * 16;
  const int lm   = lane & 15;               // batch row (B/D roles)
  const int hiB  = (lane >= 16) ? 16 : 0;   // activation-fragment K skew
  const int hi8  = (lane >= 16) ? 8  : 0;   // D-fragment hidden-row skew
  const int nbase = wave * 64;              // first hidden col of this wave

  // Per-lane swizzled weight bases: fragment block = 512 elems (32 lanes x 16).
  const __bf16* wihW = wih + ((size_t)(wave * 4) * KCE * 32 + lane) * 16;
  const __bf16* whhW = whh + ((size_t)(wave * 4) * KCH * 32 + lane) * 16;

  // h0 = 0 (buffer 0 is read at t=0)
  for (int i = tid; i < 16 * LDH; i += 256) h_lds[0][i] = (__bf16)0.f;

  // Per-lane bias vectors: acc[j][q] corresponds to hidden col
  // nbase + j*16 + hi8 + q  -> 8 consecutive floats.
  v8f bv[4];
#pragma unroll
  for (int j = 0; j < 4; ++j)
    bv[j] = *(const v8f*)(bias + nbase + j * 16 + hi8);

  const int r    = tid >> 4;   // batch row 0..15 (staging)
  const int cseg = tid & 15;   // 16-elem segment of E (staging)

  stage_xe(x, emb, xe_lds[0], b0, r, cseg, 0);
  __syncthreads();

  for (int t = 0; t < TT; ++t) {
    const int cur = t & 1;
    const int nxt = cur ^ 1;

    // ---- prestage next timestep's embeddings into the other buffer -------
    if (t + 1 < TT) {
      stage_xe(x, emb, xe_lds[nxt], b0, r, cseg, t + 1);
      if (t + 2 < TT) {  // prefetch t+2's embedding row (global_prefetch_b8)
        const int idxn = x[(b0 + r) * TT + t + 2];
        __builtin_prefetch(emb + (size_t)idxn * EE + cseg * 16, 0, 0);
      }
    }

    // ---- acc = bias -------------------------------------------------------
    v8f acc[4];
#pragma unroll
    for (int j = 0; j < 4; ++j) acc[j] = bv[j];

    // ---- input projection: acc += W_ih-tile x xe^T ------------------------
    {
      const __bf16* arow = &xe_lds[cur][lm * LDE];
#pragma unroll 4
      for (int kc = 0; kc < KCE; ++kc) {
        v16bf bact = load_act_frag(arow, kc * 32, hiB);
#pragma unroll
        for (int j = 0; j < 4; ++j) {
          v16bf aw = *(const v16bf*)(wihW + (size_t)(j * KCE + kc) * 512);
          acc[j] = __builtin_amdgcn_wmma_f32_16x16x32_bf16(
              false, aw, false, bact, (short)0, acc[j], false, false);
        }
      }
    }

    // ---- recurrence: acc += W_hh-tile x h^T -------------------------------
    {
      const __bf16* arow = &h_lds[cur][lm * LDH];
#pragma unroll 4
      for (int kc = 0; kc < KCH; ++kc) {
        v16bf bact = load_act_frag(arow, kc * 32, hiB);
#pragma unroll
        for (int j = 0; j < 4; ++j) {
          v16bf aw = *(const v16bf*)(whhW + (size_t)(j * KCH + kc) * 512);
          acc[j] = __builtin_amdgcn_wmma_f32_16x16x32_bf16(
              false, aw, false, bact, (short)0, acc[j], false, false);
        }
      }
    }

    // ---- h_next[b][n] = tanh(acc): lane lm holds batch row lm, 8 consecutive
    // hidden cols per tile -> one packed 16B LDS store per tile. -------------
#pragma unroll
    for (int j = 0; j < 4; ++j) {
      bf16x8 hv;
#pragma unroll
      for (int q = 0; q < 8; ++q) hv[q] = (__bf16)fast_tanh(acc[j][q]);
      *(bf16x8*)&h_lds[nxt][lm * LDH + nbase + j * 16 + hi8] = hv;
    }
    __syncthreads();  // one barrier per timestep
  }

  // Final hidden state is in buffer TT & 1 == 0.
  if (tid < 16 * CC) {
    const int bb = tid / CC;
    const int c  = tid % CC;
    float s = fcb[c];
    const __bf16* hrow = &h_lds[0][bb * LDH];
    const float* wrow = fcw + (size_t)c * HH;
#pragma unroll 8
    for (int k = 0; k < HH; ++k) {
      float hv = (float)hrow[k];
      hv = hv > 0.f ? hv : 0.f;
      s += hv * wrow[k];
    }
    out[(size_t)(b0 + bb) * CC + c] = s;
  }
}

// ---------------------------------------------------------------------------
extern "C" void kernel_launch(void* const* d_in, const int* in_sizes, int n_in,
                              void* d_out, int out_size, void* d_ws, size_t ws_size,
                              hipStream_t stream) {
  const int*   x   = (const int*)d_in[0];
  const float* emb = (const float*)d_in[1];
  const float* wih = (const float*)d_in[2];
  const float* whh = (const float*)d_in[3];
  const float* bih = (const float*)d_in[4];
  const float* bhh = (const float*)d_in[5];
  const float* fcw = (const float*)d_in[6];
  const float* fcb = (const float*)d_in[7];
  float* out = (float*)d_out;

  char* ws = (char*)d_ws;
  size_t off = 0;
  auto carve = [&](size_t bytes) {
    char* p = ws + off;
    off = (off + bytes + 255) & ~(size_t)255;
    return p;
  };
  __bf16* emb_bf = (__bf16*)carve((size_t)VV * EE * 2);
  __bf16* wih_sw = (__bf16*)carve((size_t)HH * EE * 2);
  __bf16* whh_sw = (__bf16*)carve((size_t)HH * HH * 2);
  float*  bias_c = (float*)carve((size_t)HH * 4);

  const int n_cvt = VV * EE;
  cvt_kernel<<<(n_cvt + 255) / 256, 256, 0, stream>>>(
      emb, wih, whh, bih, bhh, emb_bf, wih_sw, whh_sw, bias_c);

  rnn_kernel<<<BB / 16, 256, 0, stream>>>(
      x, emb_bf, wih_sw, whh_sw, bias_c, fcw, fcb, out);
}
